// SlotAttention_61074434949198
// MI455X (gfx1250) — matrix operand order repaired and checked by reading, not compiled, not run on hardware
//
#include <hip/hip_runtime.h>

#define BATCH 64
#define NTOK  4096
#define NSLOT 6
#define SLOTP 16
#define DIM   256
#define DIM2  512
#define EPS   1e-5f

typedef __attribute__((ext_vector_type(16))) __bf16 v16bf;
typedef __attribute__((ext_vector_type(8)))  float  v8f;

union BF16x16 { uint4 q[2]; v16bf v; };
union U4pack  { uint4 q; unsigned short h[8]; };

__device__ __forceinline__ unsigned short f2bf(float f) {
  union { float f; unsigned u; } x; x.f = f;
  unsigned u = x.u;
  u += 0x7FFFu + ((u >> 16) & 1u);   // round-to-nearest-even
  return (unsigned short)(u >> 16);
}

__device__ __forceinline__ v8f wmma_bf16(const BF16x16& a, const BF16x16& b, v8f c) {
  return __builtin_amdgcn_wmma_f32_16x16x32_bf16(false, a.v, false, b.v, (short)0, c,
                                                 false, false);
}

// ---------------------------------------------------------------------------
// Kernel 1: slots = mu + softplus(sigma)*noise ; zero attn_T (padding rows stay 0)
// ---------------------------------------------------------------------------
__global__ void __launch_bounds__(256) init_kernel(
    const float* __restrict__ noise, const float* __restrict__ mu,
    const float* __restrict__ sigma, float* __restrict__ slots,
    unsigned short* __restrict__ attn_T)
{
  long i = (long)blockIdx.x * 256 + threadIdx.x;
  const long totS = (long)BATCH * NSLOT * DIM;
  if (i < totS) {
    long kd = i % (NSLOT * DIM);
    float sg = sigma[kd];
    float sp = fmaxf(sg, 0.f) + log1pf(__expf(-fabsf(sg)));
    slots[i] = mu[kd] + sp * noise[i];
  }
  const long totA = (long)BATCH * SLOTP * NTOK;
  for (long j = i; j < totA; j += (long)gridDim.x * 256) attn_T[j] = 0;
}

// ---------------------------------------------------------------------------
// Kernel 2: convert Wk, Wv to bf16
// ---------------------------------------------------------------------------
__global__ void __launch_bounds__(256) wcvt_kernel(
    const float* __restrict__ Wk, const float* __restrict__ Wv,
    unsigned short* __restrict__ Wk_bf, unsigned short* __restrict__ Wv_bf)
{
  int i = blockIdx.x * 256 + threadIdx.x;
  if (i < DIM * DIM) { Wk_bf[i] = f2bf(Wk[i]); Wv_bf[i] = f2bf(Wv[i]); }
}

// ---------------------------------------------------------------------------
// Kernel 3: fused LN(x) + k = xn@Wk^T + bk, v = xn@Wv^T + bv  (WMMA bf16)
// 16 token-rows per block, 4 waves; k stored [B,N,D] bf16, v stored [B,D,N] bf16
// ---------------------------------------------------------------------------
__global__ void __launch_bounds__(128) ln_kv_kernel(
    const float* __restrict__ x, const float* __restrict__ lng,
    const float* __restrict__ lnb, const float* __restrict__ bk,
    const float* __restrict__ bv,
    const unsigned short* __restrict__ Wk_bf,
    const unsigned short* __restrict__ Wv_bf,
    unsigned short* __restrict__ k_bf, unsigned short* __restrict__ vT_bf)
{
  __shared__ float xs[16][DIM + 1];
  __shared__ __align__(16) unsigned short xbf[16][DIM + 8];
  __shared__ float redS[16][8], redQ[16][8];
  __shared__ float mu_s[16], ri_s[16];

  const int t = threadIdx.x;
  const long rowbase = (long)blockIdx.x * 16;   // flat over B*N

  for (int i = 0; i < 32; ++i) {
    int idx = t + i * 128;
    xs[idx >> 8][idx & 255] = x[(rowbase + (idx >> 8)) * DIM + (idx & 255)];
  }
  __syncthreads();
  { // stats: 8 threads per row
    int r = t >> 3, sub = t & 7;
    float s = 0.f, s2 = 0.f;
    for (int c = sub * 32; c < sub * 32 + 32; ++c) { float v = xs[r][c]; s += v; s2 += v * v; }
    redS[r][sub] = s; redQ[r][sub] = s2;
  }
  __syncthreads();
  if (t < 16) {
    float s = 0.f, s2 = 0.f;
    for (int j = 0; j < 8; ++j) { s += redS[t][j]; s2 += redQ[t][j]; }
    float mu = s * (1.f / DIM);
    mu_s[t] = mu;
    ri_s[t] = rsqrtf(s2 * (1.f / DIM) - mu * mu + EPS);
  }
  __syncthreads();
  for (int i = 0; i < 32; ++i) {
    int idx = t + i * 128; int r = idx >> 8, c = idx & 255;
    float v = (xs[r][c] - mu_s[r]) * ri_s[r] * lng[c] + lnb[c];
    xbf[r][c] = f2bf(v);
  }
  __syncthreads();

  const int wave = t >> 5, lane = t & 31;
  const int lhalf = lane >> 4, lmod = lane & 15;
  const long b  = rowbase / NTOK;
  const long nl = rowbase % NTOK;

  for (int ni = 0; ni < 4; ++ni) {
    const int nt = wave + ni * 4;            // 16 output-col tiles
    v8f ack = {}, acv = {};
    const int orow = nt * 16 + lmod;
    for (int ks = 0; ks < 8; ++ks) {
      BF16x16 a, wk, wv;
      const int kb = ks * 32 + lhalf * 16;   // A: one 16-elem run
      a.q[0] = *(const uint4*)&xbf[lmod][kb];
      a.q[1] = *(const uint4*)&xbf[lmod][kb + 8];
      const int db = ks * 32 + lhalf * 16;   // B: one 16-elem run
      const unsigned short* wkp = Wk_bf + (long)orow * DIM + db;
      const unsigned short* wvp = Wv_bf + (long)orow * DIM + db;
      wk.q[0] = *(const uint4*)wkp; wk.q[1] = *(const uint4*)(wkp + 8);
      wv.q[0] = *(const uint4*)wvp; wv.q[1] = *(const uint4*)(wvp + 8);
      ack = wmma_bf16(a, wk, ack);
      acv = wmma_bf16(a, wv, acv);
    }
    const float bkb = bk[orow], bvb = bv[orow];
    // k: [B,N,D] row-major, scattered 2B stores per C row
    for (int r = 0; r < 8; ++r) {
      int m = r + lhalf * 8;
      k_bf[(rowbase + m) * DIM + orow] = f2bf(ack[r] + bkb);
    }
    // v transposed: [B,D,N]; C rows are 8 consecutive n -> one 16B store
    U4pack pv;
    for (int r = 0; r < 8; ++r) pv.h[r] = f2bf(acv[r] + bvb);
    *(uint4*)&vT_bf[((b * DIM + orow) * NTOK) + nl + lhalf * 8] = pv.q;
  }
}

// ---------------------------------------------------------------------------
// Kernel 4 (per iter): sn = LN(slots); q = sn@Wq^T + bq -> q_bf [B,16,D], rows 6..15 = 0
// ---------------------------------------------------------------------------
__global__ void __launch_bounds__(256) slots_q_kernel(
    const float* __restrict__ slots, const float* __restrict__ lng,
    const float* __restrict__ lnb, const float* __restrict__ Wq,
    const float* __restrict__ bq, unsigned short* __restrict__ q_bf)
{
  __shared__ float sn[NSLOT][DIM];
  __shared__ float mu_s[NSLOT], ri_s[NSLOT];
  const int b = blockIdx.x, t = threadIdx.x;
  for (int i = t; i < NSLOT * DIM; i += 256)
    sn[i / DIM][i % DIM] = slots[(long)b * NSLOT * DIM + i];
  __syncthreads();
  if (t < NSLOT) {
    float s = 0.f, s2 = 0.f;
    for (int c = 0; c < DIM; ++c) { float v = sn[t][c]; s += v; s2 += v * v; }
    float mu = s * (1.f / DIM);
    mu_s[t] = mu; ri_s[t] = rsqrtf(s2 * (1.f / DIM) - mu * mu + EPS);
  }
  __syncthreads();
  for (int i = t; i < NSLOT * DIM; i += 256) {
    int s = i / DIM, c = i % DIM;
    sn[s][c] = (sn[s][c] - mu_s[s]) * ri_s[s] * lng[c] + lnb[c];
  }
  __syncthreads();
  for (int i = t; i < SLOTP * DIM; i += 256) {
    int s = i / DIM, o = i % DIM;
    unsigned short out = 0;
    if (s < NSLOT) {
      float acc = bq[o];
      const float* w = Wq + (long)o * DIM;
      for (int c = 0; c < DIM; ++c) acc += sn[s][c] * w[c];
      out = f2bf(acc);
    }
    q_bf[((long)b * SLOTP + s) * DIM + o] = out;
  }
}

// ---------------------------------------------------------------------------
// Kernel 5 (per iter): logits = (k @ q^T) * D^-0.5 ; softmax over 6 slots ->
// attn_T [B,16,N] bf16 (rows 6..15 stay zero). 8 waves x 16 rows per block.
// ---------------------------------------------------------------------------
__global__ void __launch_bounds__(256) logits_attn_kernel(
    const unsigned short* __restrict__ k_bf,
    const unsigned short* __restrict__ q_bf,
    unsigned short* __restrict__ attn_T)
{
  __shared__ float lt[8][16][17];
  const int t = threadIdx.x, wave = t >> 5, lane = t & 31;
  const int lhalf = lane >> 4, lmod = lane & 15;
  const int b    = blockIdx.x >> 5;          // N/128 = 32 blocks per batch
  const int nblk = blockIdx.x & 31;
  const long rowbase = (long)b * NTOK + nblk * 128 + wave * 16;

  v8f acc = {};
  for (int ks = 0; ks < 8; ++ks) {
    BF16x16 a, bm;
    const unsigned short* kp = k_bf + (rowbase + lmod) * DIM + ks * 32 + lhalf * 16;
    a.q[0] = *(const uint4*)kp;
    a.q[1] = *(const uint4*)(kp + 8);
    const unsigned short* qp = q_bf + ((long)b * SLOTP + lmod) * DIM + ks * 32 + lhalf * 16;
    bm.q[0] = *(const uint4*)qp;
    bm.q[1] = *(const uint4*)(qp + 8);
    acc = wmma_bf16(a, bm, acc);
  }
  const float scale = 0.0625f;               // D^-0.5 = 1/16
  for (int r = 0; r < 8; ++r) lt[wave][r + lhalf * 8][lmod] = acc[r] * scale;
  __syncthreads();
  if (t < 128) {
    int wv = t >> 4, m = t & 15;
    float mx = lt[wv][m][0];
    for (int s = 1; s < NSLOT; ++s) mx = fmaxf(mx, lt[wv][m][s]);
    float e[NSLOT], sum = 0.f;
    for (int s = 0; s < NSLOT; ++s) { e[s] = __expf(lt[wv][m][s] - mx); sum += e[s]; }
    float inv = 1.f / sum;
    long n = (long)nblk * 128 + wv * 16 + m;
    for (int s = 0; s < NSLOT; ++s)
      attn_T[((long)b * SLOTP + s) * NTOK + n] = f2bf(e[s] * inv);
  }
}

// ---------------------------------------------------------------------------
// Kernel 6 (per iter): updates = attn^T @ v  ([16,N]x[N,D] via WMMA, K=N=4096)
// 4 waves/block, one 16-col d-tile each; vT gives contiguous B-operand runs.
// ---------------------------------------------------------------------------
__global__ void __launch_bounds__(128) updates_kernel(
    const unsigned short* __restrict__ attn_T,
    const unsigned short* __restrict__ vT_bf,
    float* __restrict__ updates)
{
  const int t = threadIdx.x, wave = t >> 5, lane = t & 31;
  const int lhalf = lane >> 4, lmod = lane & 15;
  const int b  = blockIdx.x >> 2;
  const int dt = (blockIdx.x & 3) * 4 + wave;      // 0..15 d-tiles
  const unsigned short* ap = attn_T + ((long)b * SLOTP + lmod) * NTOK;
  const unsigned short* vp = vT_bf + ((long)b * DIM + dt * 16 + lmod) * NTOK;

  v8f acc0 = {}, acc1 = {};
  for (int kc = 0; kc < 128; kc += 2) {
    BF16x16 a, bm;
    int nb = kc * 32;
    const unsigned short* aa = ap + nb + lhalf * 16;
    a.q[0] = *(const uint4*)aa; a.q[1] = *(const uint4*)(aa + 8);
    const unsigned short* bb = vp + nb + lhalf * 16;
    bm.q[0] = *(const uint4*)bb; bm.q[1] = *(const uint4*)(bb + 8);
    acc0 = wmma_bf16(a, bm, acc0);
    nb += 32;
    aa = ap + nb + lhalf * 16;
    a.q[0] = *(const uint4*)aa; a.q[1] = *(const uint4*)(aa + 8);
    bb = vp + nb + lhalf * 16;
    bm.q[0] = *(const uint4*)bb; bm.q[1] = *(const uint4*)(bb + 8);
    acc1 = wmma_bf16(a, bm, acc1);
  }
  const int d = dt * 16 + lmod;
  for (int r = 0; r < 8; ++r) {
    int s = r + lhalf * 8;
    updates[((long)b * SLOTP + s) * DIM + d] = acc0[r] + acc1[r];
  }
}

// ---------------------------------------------------------------------------
// Kernel 7 (per iter): GRU cell + LN + MLP residual, fp32, in-place on slots
// ---------------------------------------------------------------------------
__global__ void __launch_bounds__(256) gru_mlp_kernel(
    const float* __restrict__ updates, float* __restrict__ slots,
    const float* __restrict__ W_ih, const float* __restrict__ b_ih,
    const float* __restrict__ W_hh, const float* __restrict__ b_hh,
    const float* __restrict__ W1, const float* __restrict__ b1,
    const float* __restrict__ W2, const float* __restrict__ b2,
    const float* __restrict__ lng, const float* __restrict__ lnb)
{
  __shared__ float up[NSLOT][DIM], sp[NSLOT][DIM], ns[NSLOT][DIM];
  __shared__ float act[NSLOT][DIM2];
  __shared__ float mu_s[NSLOT], ri_s[NSLOT];
  const int b = blockIdx.x, t = threadIdx.x;

  for (int i = t; i < NSLOT * DIM; i += 256) {
    int s = i / DIM, d = i % DIM;
    up[s][d] = updates[((long)b * SLOTP + s) * DIM + d];
    sp[s][d] = slots[(long)b * NSLOT * DIM + i];
  }
  __syncthreads();
  for (int i = t; i < NSLOT * DIM; i += 256) {
    int s = i / DIM, d = i % DIM;
    float gir = b_ih[d], giz = b_ih[DIM + d], gin = b_ih[2 * DIM + d];
    float ghr = b_hh[d], ghz = b_hh[DIM + d], ghn = b_hh[2 * DIM + d];
    const float* wir = W_ih + (long)d * DIM;
    const float* wiz = W_ih + (long)(DIM + d) * DIM;
    const float* win = W_ih + (long)(2 * DIM + d) * DIM;
    const float* whr = W_hh + (long)d * DIM;
    const float* whz = W_hh + (long)(DIM + d) * DIM;
    const float* whn = W_hh + (long)(2 * DIM + d) * DIM;
    for (int c = 0; c < DIM; ++c) {
      float u = up[s][c], h = sp[s][c];
      gir += u * wir[c]; giz += u * wiz[c]; gin += u * win[c];
      ghr += h * whr[c]; ghz += h * whz[c]; ghn += h * whn[c];
    }
    float r = 1.f / (1.f + __expf(-(gir + ghr)));
    float z = 1.f / (1.f + __expf(-(giz + ghz)));
    float nn = tanhf(gin + r * ghn);
    ns[s][d] = (1.f - z) * nn + z * sp[s][d];
  }
  __syncthreads();
  if (t < NSLOT) {
    float s = 0.f, s2 = 0.f;
    for (int c = 0; c < DIM; ++c) { float v = ns[t][c]; s += v; s2 += v * v; }
    float mu = s * (1.f / DIM);
    mu_s[t] = mu; ri_s[t] = rsqrtf(s2 * (1.f / DIM) - mu * mu + EPS);
  }
  __syncthreads();
  for (int i = t; i < NSLOT * DIM; i += 256) {   // h = LN(ns), reuse sp
    int s = i / DIM, d = i % DIM;
    sp[s][d] = (ns[s][d] - mu_s[s]) * ri_s[s] * lng[d] + lnb[d];
  }
  __syncthreads();
  for (int i = t; i < NSLOT * DIM2; i += 256) {
    int s = i / DIM2, o = i % DIM2;
    float a = b1[o];
    const float* w = W1 + (long)o * DIM;
    for (int c = 0; c < DIM; ++c) a += sp[s][c] * w[c];
    act[s][o] = fmaxf(a, 0.f);
  }
  __syncthreads();
  for (int i = t; i < NSLOT * DIM; i += 256) {
    int s = i / DIM, d = i % DIM;
    float a = b2[d];
    const float* w = W2 + (long)d * DIM2;
    for (int c = 0; c < DIM2; ++c) a += act[s][c] * w[c];
    slots[(long)b * NSLOT * DIM + i] = ns[s][d] + a;
  }
}

// ---------------------------------------------------------------------------
extern "C" void kernel_launch(void* const* d_in, const int* in_sizes, int n_in,
                              void* d_out, int out_size, void* d_ws, size_t ws_size,
                              hipStream_t stream)
{
  const float* x        = (const float*)d_in[0];
  const float* noise    = (const float*)d_in[1];
  const float* slots_mu = (const float*)d_in[2];
  const float* slots_sg = (const float*)d_in[3];
  const float* Wq  = (const float*)d_in[4];
  const float* bq  = (const float*)d_in[5];
  const float* Wk  = (const float*)d_in[6];
  const float* bk  = (const float*)d_in[7];
  const float* Wv  = (const float*)d_in[8];
  const float* bv  = (const float*)d_in[9];
  const float* W_ih = (const float*)d_in[10];
  const float* b_ih = (const float*)d_in[11];
  const float* W_hh = (const float*)d_in[12];
  const float* b_hh = (const float*)d_in[13];
  const float* W1  = (const float*)d_in[14];
  const float* b1  = (const float*)d_in[15];
  const float* W2  = (const float*)d_in[16];
  const float* b2  = (const float*)d_in[17];
  const float* ln_in_g = (const float*)d_in[18];
  const float* ln_in_b = (const float*)d_in[19];
  const float* ln_s_g  = (const float*)d_in[20];
  const float* ln_s_b  = (const float*)d_in[21];
  const float* ln_f_g  = (const float*)d_in[22];
  const float* ln_f_b  = (const float*)d_in[23];
  float* slots = (float*)d_out;
  (void)in_sizes; (void)n_in; (void)out_size; (void)ws_size;

  char* ws = (char*)d_ws;
  size_t off = 0;
  auto carve = [&](size_t bytes) -> char* {
    char* p = ws + off;
    off = (off + bytes + 255) & ~(size_t)255;
    return p;
  };
  unsigned short* Wk_bf  = (unsigned short*)carve((size_t)DIM * DIM * 2);
  unsigned short* Wv_bf  = (unsigned short*)carve((size_t)DIM * DIM * 2);
  unsigned short* q_bf   = (unsigned short*)carve((size_t)BATCH * SLOTP * DIM * 2);
  unsigned short* attn_T = (unsigned short*)carve((size_t)BATCH * SLOTP * NTOK * 2);
  float*          upd    = (float*)carve((size_t)BATCH * SLOTP * DIM * 4);
  unsigned short* k_bf   = (unsigned short*)carve((size_t)BATCH * NTOK * DIM * 2);
  unsigned short* vT_bf  = (unsigned short*)carve((size_t)BATCH * NTOK * DIM * 2);

  init_kernel<<<2048, 256, 0, stream>>>(noise, slots_mu, slots_sg, slots, attn_T);
  wcvt_kernel<<<(DIM * DIM + 255) / 256, 256, 0, stream>>>(Wk, Wv, Wk_bf, Wv_bf);
  ln_kv_kernel<<<(BATCH * NTOK) / 16, 128, 0, stream>>>(
      x, ln_in_g, ln_in_b, bk, bv, Wk_bf, Wv_bf, k_bf, vT_bf);

  for (int it = 0; it < 3; ++it) {
    slots_q_kernel<<<BATCH, 256, 0, stream>>>(slots, ln_s_g, ln_s_b, Wq, bq, q_bf);
    logits_attn_kernel<<<BATCH * (NTOK / 128), 256, 0, stream>>>(k_bf, q_bf, attn_T);
    updates_kernel<<<BATCH * 4, 128, 0, stream>>>(attn_T, vT_bf, upd);
    gru_mlp_kernel<<<BATCH, 256, 0, stream>>>(upd, slots, W_ih, b_ih, W_hh, b_hh,
                                              W1, b1, W2, b2, ln_f_g, ln_f_b);
  }
}
